// Mamba2Block_1889785610994
// MI455X (gfx1250) — matrix operand order repaired and checked by reading, not compiled, not run on hardware
//
#include <hip/hip_runtime.h>
#include <hip/hip_bf16.h>
#include <math.h>

// ---------------------------------------------------------------------------
// Mamba block forward for MI455X (gfx1250), wave32.
// WMMA bf16 GEMMs with async-to-LDS double-buffered B staging.
// ---------------------------------------------------------------------------

typedef __bf16 bf16_t;
typedef __attribute__((ext_vector_type(8)))  __bf16 v8bf;
typedef __attribute__((ext_vector_type(16))) __bf16 v16bf;
typedef __attribute__((ext_vector_type(8)))  float  v8f;

#define H_DIM   1024
#define D_IN    2048
#define N_STATE 16
#define R_RANK  128
#define SEQ_L   2048
#define BATCH   2
#define TOKENS  (BATCH * SEQ_L)   // 4096

// ---------------------------------------------------------------------------
// fp32 -> bf16 conversion (weights)
// ---------------------------------------------------------------------------
__global__ void f32_to_bf16_kernel(const float* __restrict__ src,
                                   bf16_t* __restrict__ dst, int n) {
    int i = blockIdx.x * blockDim.x + threadIdx.x;
    if (i < n) dst[i] = (bf16_t)src[i];
}

// ---------------------------------------------------------------------------
// RMSNorm over H=1024 per token, output bf16. One block (256 thr) per token.
// ---------------------------------------------------------------------------
__global__ void rmsnorm_kernel(const float* __restrict__ x,
                               const float* __restrict__ w,
                               bf16_t* __restrict__ xn) {
    const int t   = blockIdx.x;
    const int tid = threadIdx.x;
    __shared__ float red[256];
    const float* row = x + (size_t)t * H_DIM;
    float s = 0.f;
    #pragma unroll
    for (int j = tid; j < H_DIM; j += 256) { float v = row[j]; s += v * v; }
    red[tid] = s;
    __syncthreads();
    for (int off = 128; off > 0; off >>= 1) {
        if (tid < off) red[tid] += red[tid + off];
        __syncthreads();
    }
    const float rs = rsqrtf(red[0] * (1.0f / H_DIM) + 1e-6f);
    #pragma unroll
    for (int j = tid; j < H_DIM; j += 256)
        xn[(size_t)t * H_DIM + j] = (bf16_t)(row[j] * rs * w[j]);
}

// ---------------------------------------------------------------------------
// WMMA bf16 GEMM: C[M,N] = A[M,K] @ W[N,K]^T  (both row-major, K innermost).
// Block = 256 threads = 8 waves. All waves share the N-range (16*NT cols),
// wave w owns M-tile blockIdx.y*8+w.
//   - B tile (NT*16 rows x 32 k, bf16) staged in LDS with
//     GLOBAL_LOAD_ASYNC_TO_LDS (double buffered, ASYNCcnt-tracked).
//   - A fragments loaded per-wave from global with register double-buffering.
// Fragment layouts per CDNA5 ISA 7.12.2 (wave32).
// EPI: 0 = store; 1 = store + bf16 copy of first 128 cols (x_proj);
//      2 = +bias, softplus (dt_proj); 3 = +residual (out_proj).
// ---------------------------------------------------------------------------
template <int NT, int EPI>
__global__ void gemm_wmma_kernel(const bf16_t* __restrict__ A,
                                 const bf16_t* __restrict__ W,
                                 float* __restrict__ C,
                                 int M, int N, int K,
                                 const float* __restrict__ bias,
                                 const float* __restrict__ resid,
                                 bf16_t* __restrict__ out2) {
    const int tid   = threadIdx.x;
    const int lane  = tid & 31;
    const int wave  = tid >> 5;
    const int mtile = blockIdx.y * 8 + wave;
    const int nbase = blockIdx.x * (16 * NT);
    const int half  = lane >> 4;          // 0: lanes 0-15, 1: lanes 16-31
    const int l16   = lane & 15;

    // B staging buffers: [2][NT*16 rows][32 k] bf16, row = 64 bytes.
    __shared__ __align__(16) bf16_t sB[2][NT * 16 * 32];

    // Cooperative async-copy assignment: NT*1024 bytes per k-step, 256 thr.
    size_t   g_rowoff;   // element offset into W (without kt)
    unsigned lds_off;    // byte offset into one buffer
    if (NT == 4) {       // 16 B per thread
        const int row = tid >> 2, seg = tid & 3;
        g_rowoff = (size_t)(nbase + row) * K + seg * 8;
        lds_off  = row * 64 + seg * 16;
    } else {             // NT == 1: 4 B per thread
        const int row = tid >> 4, seg = tid & 15;
        g_rowoff = (size_t)(nbase + row) * K + seg * 2;
        lds_off  = row * 64 + seg * 4;
    }
    const unsigned ldsbase0 = (unsigned)(size_t)(&sB[0][0]);
    const unsigned ldsbase1 = (unsigned)(size_t)(&sB[1][0]);

    const bf16_t* Arow = A + (size_t)(mtile * 16 + l16) * K;

    v8f acc[NT] = {};

    // ---- prologue: issue async B copy for kt = 0 into buffer 0, load A(0).
    {
        const bf16_t* gaddr = W + g_rowoff;
        const unsigned laddr = ldsbase0 + lds_off;
        if (NT == 4)
            asm volatile("global_load_async_to_lds_b128 %0, %1, off"
                         :: "v"(laddr), "v"(gaddr) : "memory");
        else
            asm volatile("global_load_async_to_lds_b32 %0, %1, off"
                         :: "v"(laddr), "v"(gaddr) : "memory");
    }
    v8bf alo = *(const v8bf*)(Arow + half * 8);
    v8bf ahi = *(const v8bf*)(Arow + 16 + half * 8);

    int buf = 0;
    for (int kt = 0; kt < K; kt += 32) {
        const int ktn = kt + 32;
        v8bf nlo{}, nhi{};
        if (ktn < K) {
            // issue async copy of next B tile into the other buffer
            const bf16_t* gaddr = W + g_rowoff + ktn;
            const unsigned laddr = (buf ? ldsbase0 : ldsbase1) + lds_off;
            if (NT == 4)
                asm volatile("global_load_async_to_lds_b128 %0, %1, off"
                             :: "v"(laddr), "v"(gaddr) : "memory");
            else
                asm volatile("global_load_async_to_lds_b32 %0, %1, off"
                             :: "v"(laddr), "v"(gaddr) : "memory");
            // prefetch next A fragment into registers
            nlo = *(const v8bf*)(Arow + ktn + half * 8);
            nhi = *(const v8bf*)(Arow + ktn + 16 + half * 8);
            // wait for the *current* buffer's copy (1 outstanding allowed)
            asm volatile("s_wait_asynccnt 0x1" ::: "memory");
        } else {
            asm volatile("s_wait_asynccnt 0x0" ::: "memory");
        }
        __syncthreads();   // all waves' copies of current buffer complete

        v16bf afrag = __builtin_shufflevector(alo, ahi,
            0, 1, 2, 3, 4, 5, 6, 7, 8, 9, 10, 11, 12, 13, 14, 15);

        const bf16_t* sbuf = &sB[buf][0];
        #pragma unroll
        for (int j = 0; j < NT; ++j) {
            // B fragment 32x16 from LDS: lane n = l16, k = half*8 + {0..7}, 16 + half*8 + {0..7}
            const bf16_t* Brow = sbuf + (j * 16 + l16) * 32 + half * 8;
            v8bf blo = *(const v8bf*)(Brow);
            v8bf bhi = *(const v8bf*)(Brow + 16);
            v16bf bfrag = __builtin_shufflevector(blo, bhi,
                0, 1, 2, 3, 4, 5, 6, 7, 8, 9, 10, 11, 12, 13, 14, 15);
            acc[j] = __builtin_amdgcn_wmma_f32_16x16x32_bf16(
                false, afrag, false, bfrag, (short)0, acc[j], false, false);
        }
        __syncthreads();   // done reading current buffer
        alo = nlo; ahi = nhi;
        buf ^= 1;
    }

    // C/D layout: VGPR r, lane -> (m = r + 8*half, n = l16)
    #pragma unroll
    for (int j = 0; j < NT; ++j) {
        #pragma unroll
        for (int r = 0; r < 8; ++r) {
            const int row = mtile * 16 + r + half * 8;
            const int col = nbase + j * 16 + l16;
            const size_t idx = (size_t)row * N + col;
            float v = acc[j][r];
            if (EPI == 0) {
                C[idx] = v;
            } else if (EPI == 1) {
                C[idx] = v;
                if (col < R_RANK)
                    out2[(size_t)row * R_RANK + col] = (bf16_t)v;
            } else if (EPI == 2) {
                v += bias[col];
                v = (v > 20.f) ? v : log1pf(__expf(v));   // softplus
                C[idx] = v;
            } else {
                C[idx] = v + resid[idx];
            }
        }
    }
}

// ---------------------------------------------------------------------------
// Depthwise causal conv (K=4) + bias + SiLU over x-half of xz.
// xz layout: [token, 4096] fp32 (cols 0..2047 = x path, 2048..4095 = z path).
// ---------------------------------------------------------------------------
__global__ void conv_silu_kernel(const float* __restrict__ xz,
                                 const float* __restrict__ conv_w,
                                 const float* __restrict__ conv_b,
                                 float* __restrict__ xp_f,
                                 bf16_t* __restrict__ xp_b) {
    const int e = blockIdx.x * blockDim.x + threadIdx.x;   // < TOKENS*D_IN
    const int d = e & (D_IN - 1);
    const int t = e >> 11;
    const int l = t & (SEQ_L - 1);
    float acc = conv_b[d];
    #pragma unroll
    for (int k = 0; k < 4; ++k) {
        const int lp = l - 3 + k;
        if (lp >= 0)
            acc += xz[(size_t)(t - 3 + k) * (2 * D_IN) + d] * conv_w[d * 4 + k];
    }
    const float s = acc / (1.f + __expf(-acc));   // SiLU
    xp_f[e] = s;
    xp_b[e] = (bf16_t)s;
}

// ---------------------------------------------------------------------------
// Selective scan, fused with SiLU(z) gating. One thread per (b, d) channel;
// Bm/Cm (32 floats per step) staged in LDS, N=16 state in registers.
// 64 blocks * 64 threads = BATCH * D_IN channels (spread across WGPs).
// ---------------------------------------------------------------------------
__global__ void scan_kernel(const float* __restrict__ xpath,
                            const float* __restrict__ dt,
                            const float* __restrict__ xdbl,   // [tok,160]
                            const float* __restrict__ xz,     // z at col 2048+d
                            const float* __restrict__ A_log,
                            const float* __restrict__ D_param,
                            bf16_t* __restrict__ yg) {
    const int tid = threadIdx.x;
    const int d   = (blockIdx.x & 31) * 64 + tid;
    const int b   = blockIdx.x >> 5;
    __shared__ float sBC[32];

    float Ac[N_STATE];
    #pragma unroll
    for (int n = 0; n < N_STATE; ++n) Ac[n] = -__expf(A_log[d * N_STATE + n]);
    const float Dv = D_param[d];
    float h[N_STATE];
    #pragma unroll
    for (int n = 0; n < N_STATE; ++n) h[n] = 0.f;

    for (int l = 0; l < SEQ_L; ++l) {
        const int t = b * SEQ_L + l;
        __syncthreads();
        if (tid < 32) sBC[tid] = xdbl[(size_t)t * (R_RANK + 2 * N_STATE) + R_RANK + tid];
        __syncthreads();
        const float dtv = dt[(size_t)t * D_IN + d];
        const float xv  = xpath[(size_t)t * D_IN + d];
        const float dx  = dtv * xv;
        float y = 0.f;
        #pragma unroll
        for (int n = 0; n < N_STATE; ++n) {
            const float dA = __expf(dtv * Ac[n]);
            h[n] = dA * h[n] + dx * sBC[n];
            y += h[n] * sBC[16 + n];
        }
        y += Dv * xv;
        const float z = xz[(size_t)t * (2 * D_IN) + D_IN + d];
        const float g = z / (1.f + __expf(-z));
        yg[(size_t)t * D_IN + d] = (bf16_t)(y * g);
    }
}

// ---------------------------------------------------------------------------
// Host launcher
// ---------------------------------------------------------------------------
extern "C" void kernel_launch(void* const* d_in, const int* in_sizes, int n_in,
                              void* d_out, int out_size, void* d_ws, size_t ws_size,
                              hipStream_t stream) {
    const float* x          = (const float*)d_in[0];   // [2,2048,1024]
    const float* norm_w     = (const float*)d_in[1];   // [1024]
    const float* in_proj_w  = (const float*)d_in[2];   // [4096,1024]
    const float* conv_w     = (const float*)d_in[3];   // [2048,1,4]
    const float* conv_b     = (const float*)d_in[4];   // [2048]
    const float* x_proj_w   = (const float*)d_in[5];   // [160,2048]
    const float* dt_proj_w  = (const float*)d_in[6];   // [2048,128]
    const float* dt_proj_b  = (const float*)d_in[7];   // [2048]
    const float* A_log      = (const float*)d_in[8];   // [2048,16]
    const float* D_param    = (const float*)d_in[9];   // [2048]
    const float* out_proj_w = (const float*)d_in[10];  // [1024,2048]
    float* out = (float*)d_out;                        // [2,2048,1024]

    char* w = (char*)d_ws;
    size_t off = 0;
    auto carve = [&](size_t bytes) -> void* {
        void* p = w + off;
        off += (bytes + 255) & ~(size_t)255;
        return p;
    };

    bf16_t* xn_bf   = (bf16_t*)carve((size_t)TOKENS * H_DIM * 2);
    bf16_t* win_bf  = (bf16_t*)carve((size_t)2 * D_IN * H_DIM * 2);
    bf16_t* wxp_bf  = (bf16_t*)carve((size_t)(R_RANK + 2 * N_STATE) * D_IN * 2);
    bf16_t* wdt_bf  = (bf16_t*)carve((size_t)D_IN * R_RANK * 2);
    bf16_t* wout_bf = (bf16_t*)carve((size_t)H_DIM * D_IN * 2);
    float*  xz      = (float*)carve((size_t)TOKENS * 2 * D_IN * 4);
    float*  xp_f    = (float*)carve((size_t)TOKENS * D_IN * 4);
    bf16_t* xp_b    = (bf16_t*)carve((size_t)TOKENS * D_IN * 2);
    float*  xdbl    = (float*)carve((size_t)TOKENS * (R_RANK + 2 * N_STATE) * 4);
    bf16_t* dtx_bf  = (bf16_t*)carve((size_t)TOKENS * R_RANK * 2);
    float*  dt_f    = (float*)carve((size_t)TOKENS * D_IN * 4);
    bf16_t* yg_bf   = (bf16_t*)carve((size_t)TOKENS * D_IN * 2);
    (void)ws_size;

    // 1) weight conversions fp32 -> bf16
    {
        int n1 = 2 * D_IN * H_DIM;
        f32_to_bf16_kernel<<<n1 / 256, 256, 0, stream>>>(in_proj_w, win_bf, n1);
        int n2 = (R_RANK + 2 * N_STATE) * D_IN;
        f32_to_bf16_kernel<<<n2 / 256, 256, 0, stream>>>(x_proj_w, wxp_bf, n2);
        int n3 = D_IN * R_RANK;
        f32_to_bf16_kernel<<<n3 / 256, 256, 0, stream>>>(dt_proj_w, wdt_bf, n3);
        int n4 = H_DIM * D_IN;
        f32_to_bf16_kernel<<<n4 / 256, 256, 0, stream>>>(out_proj_w, wout_bf, n4);
    }

    // 2) RMSNorm -> bf16 activations
    rmsnorm_kernel<<<TOKENS, 256, 0, stream>>>(x, norm_w, xn_bf);

    // 3) in_proj GEMM: xz[4096,4096] = xn @ in_proj_w^T
    {
        dim3 grid((2 * D_IN) / 64, TOKENS / 128);
        gemm_wmma_kernel<4, 0><<<grid, 256, 0, stream>>>(
            xn_bf, win_bf, xz, TOKENS, 2 * D_IN, H_DIM, nullptr, nullptr, nullptr);
    }

    // 4) depthwise causal conv + SiLU
    conv_silu_kernel<<<(TOKENS * D_IN) / 256, 256, 0, stream>>>(
        xz, conv_w, conv_b, xp_f, xp_b);

    // 5) x_proj GEMM: xdbl[4096,160] = xpath @ x_proj_w^T (+ bf16 dt_x copy)
    {
        dim3 grid((R_RANK + 2 * N_STATE) / 16, TOKENS / 128);
        gemm_wmma_kernel<1, 1><<<grid, 256, 0, stream>>>(
            xp_b, wxp_bf, xdbl, TOKENS, R_RANK + 2 * N_STATE, D_IN,
            nullptr, nullptr, dtx_bf);
    }

    // 6) dt_proj GEMM: dt[4096,2048] = softplus(dt_x @ dt_proj_w^T + b)
    {
        dim3 grid(D_IN / 64, TOKENS / 128);
        gemm_wmma_kernel<4, 2><<<grid, 256, 0, stream>>>(
            dtx_bf, wdt_bf, dt_f, TOKENS, D_IN, R_RANK,
            dt_proj_b, nullptr, nullptr);
    }

    // 7) selective scan + SiLU(z) gating -> bf16 y
    scan_kernel<<<BATCH * (D_IN / 64), 64, 0, stream>>>(
        xp_f, dt_f, xdbl, xz, A_log, D_param, yg_bf);

    // 8) out_proj GEMM with residual: out[4096,1024] = y @ out_proj_w^T + x
    {
        dim3 grid(H_DIM / 64, TOKENS / 128);
        gemm_wmma_kernel<4, 3><<<grid, 256, 0, stream>>>(
            yg_bf, wout_bf, out, TOKENS, H_DIM, D_IN,
            nullptr, x, nullptr);
    }
    (void)in_sizes; (void)n_in; (void)out_size;
}
